// RQVAEGatedDual_4140348473616
// MI455X (gfx1250) — matrix operand order repaired and checked by reading, not compiled, not run on hardware
//
#include <hip/hip_runtime.h>
#include <hip/hip_bf16.h>

typedef __attribute__((ext_vector_type(16))) _Float16 v16h;
typedef __attribute__((ext_vector_type(8)))  float    v8f;
typedef __attribute__((ext_vector_type(4)))  float    f32x4;

#define NWAVES 4
static constexpr int B_ROWS  = 65536;
static constexpr int PITCH_A = 1544;   // holds up to 1536 cols, 16B-aligned rows
static constexpr int PITCH_B = 848;    // holds up to 840 cols

// ---- output layout (flat float slots; codes stored as int32 bits) ----
static constexpr long long O_SEM   = 0;
static constexpr long long O_COL   = (long long)B_ROWS * 768;
static constexpr long long O_ZQ    = O_COL + (long long)B_ROWS * 64;
static constexpr long long O_CODES = O_ZQ  + (long long)B_ROWS * 64;
static constexpr long long O_LOSS  = O_CODES + (long long)B_ROWS * 3;
static constexpr long long O_GATE  = O_LOSS + 2;

// ---- workspace layout (halves) ----
static constexpr size_t W_GATE1 = 0;
static constexpr size_t W_GATE2 = W_GATE1 + 64 * 128;
static constexpr size_t W_ENC1  = W_GATE2 + 128 * 64;
static constexpr size_t W_ENC2  = W_ENC1  + 832 * 512;
static constexpr size_t W_ENC3  = W_ENC2  + 512 * 256;
static constexpr size_t W_ENC4  = W_ENC3  + 256 * 128;
static constexpr size_t W_DEC1  = W_ENC4  + 128 * 64;
static constexpr size_t W_DEC2  = W_DEC1  + 64 * 128;
static constexpr size_t W_DEC3  = W_DEC2  + 128 * 256;
static constexpr size_t W_SEM1  = W_DEC3  + 256 * 512;
static constexpr size_t W_SEM2  = W_SEM1  + 512 * 1536;
static constexpr size_t W_COL1  = W_SEM2  + 1536 * 768;
static constexpr size_t W_COL2  = W_COL1  + 512 * 256;
static constexpr size_t W_CBT   = W_COL2  + 256 * 64;       // cb^T : [3][64][256]
static constexpr size_t W_ENDH  = W_CBT   + 3 * 64 * 256;
static constexpr size_t EH_BYTE = ((W_ENDH * 2 + 255) / 256) * 256;  // 768 f32 after

struct SMem {
  alignas(16) _Float16 bufA[16][PITCH_A];
  alignas(16) _Float16 bufB[16][PITCH_B];
  float small[16][68];     // collab f32, later residual (f32)
  float resn [16][68];     // l2-normalized residual
  float zq   [16][68];     // running z_q_total
  float red  [16][8][2];   // row reduction scratch
  float mstats[16][2];
  float candV[NWAVES][16];
  int   candI[NWAVES][16];
  int   codes[16];
  float loss;
};

// ---------- WMMA fragment loaders (CDNA5 wave32 layouts) ----------
// A (16x32 f16): lanes 0-15 = M, lanes 16-31 same M other K-half.
// VGPR i<4 -> K = 2i..2i+1 (+8*half); i>=4 -> K = 2i+8.. (+8*half)
__device__ __forceinline__ v16h ld_a(const _Float16* base, int pitch, int kBase, int lane) {
  int m = lane & 15, hf = lane >> 4;
  const _Float16* p = base + m * pitch + kBase + 8 * hf;
  union { v16h v; f32x4 q[2]; } u;
  u.q[0] = *reinterpret_cast<const f32x4*>(p);        // K rel 0..7
  u.q[1] = *reinterpret_cast<const f32x4*>(p + 16);   // K rel 16..23
  return u.v;
}
// B (32x16 f16), W row-major [K][N]: lane holds column n = nBase+lane%16, K in same order as A
__device__ __forceinline__ v16h ld_b(const _Float16* __restrict__ W, int Ndim,
                                     int kBase, int nBase, int lane) {
  int n = lane & 15, hf = lane >> 4;
  const _Float16* p = W + (size_t)kBase * Ndim + nBase + n;
  v16h v;
#pragma unroll
  for (int s = 0; s < 16; ++s) {
    int k = (s < 8) ? (8 * hf + s) : (16 + 8 * hf + (s - 8));
    v[s] = p[(size_t)k * Ndim];
  }
  return v;
}

__device__ __forceinline__ float silu_f(float v) { return v / (1.0f + __expf(-v)); }

// Dual N-tile GEMM: each wave owns pairs of 16-col tiles (Ndim must be multiple of 32).
// A fragment loaded once per K-step, feeds two independent WMMAs (no D->A/B hazard).
// modes: 0 = +bias -> f16 LDS ; 1 = silu -> f16 LDS ; 3 = +bias -> f32 global ; 4 = +bias -> f32 LDS
__device__ void gemm_stage(const _Float16* aBase, int aPitch, int Kdim,
                           const _Float16* __restrict__ W, int Ndim,
                           const float* __restrict__ bias, int mode,
                           _Float16* oLds, int oPitch,
                           float* oF32, int oFPitch,
                           float* oG, int oGPitch, long long rowBase,
                           int waveId, int lane) {
  int nPairs = Ndim >> 5;
  for (int p = waveId; p < nPairs; p += NWAVES) {
    int nB0 = p << 5, nB1 = nB0 + 16;
    v8f c0 = {}, c1 = {};
    for (int kb = 0; kb < Kdim; kb += 32) {
      v16h a  = ld_a(aBase, aPitch, kb, lane);
      v16h b0 = ld_b(W, Ndim, kb, nB0, lane);
      v16h b1 = ld_b(W, Ndim, kb, nB1, lane);
      c0 = __builtin_amdgcn_wmma_f32_16x16x32_f16(false, a, false, b0, (short)0, c0, false, false);
      c1 = __builtin_amdgcn_wmma_f32_16x16x32_f16(false, a, false, b1, (short)0, c1, false, false);
    }
    int n = lane & 15, hf = lane >> 4;
    float bv0 = bias ? bias[nB0 + n] : 0.0f;
    float bv1 = bias ? bias[nB1 + n] : 0.0f;
#pragma unroll
    for (int r = 0; r < 8; ++r) {
      int m = r + (hf << 3);
      float v0 = c0[r] + bv0;
      float v1 = c1[r] + bv1;
      if (mode == 1) { v0 = silu_f(v0); v1 = silu_f(v1); }
      if (mode == 3) {
        oG[(rowBase + m) * (long long)oGPitch + nB0 + n] = v0;
        oG[(rowBase + m) * (long long)oGPitch + nB1 + n] = v1;
      } else if (mode == 4) {
        oF32[m * oFPitch + nB0 + n] = v0;
        oF32[m * oFPitch + nB1 + n] = v1;
      } else {
        oLds[m * oPitch + nB0 + n] = (_Float16)v0;
        oLds[m * oPitch + nB1 + n] = (_Float16)v1;
      }
    }
  }
}

// Row LayerNorm over `dim` (multiple of 8), 128 threads, optional ReLU, in==out OK.
__device__ void ln_block(SMem& sh, const _Float16* in, int inPitch, int dim,
                         const float* __restrict__ g, const float* __restrict__ bta,
                         _Float16* out, int outPitch, bool relu, int tid) {
  int row = tid & 15, ch = tid >> 4;
  int per = dim >> 3;
  const _Float16* ip = in + row * inPitch;
  float s = 0.f, s2 = 0.f;
  for (int j = ch * per; j < (ch + 1) * per; ++j) {
    float x = (float)ip[j]; s += x; s2 += x * x;
  }
  sh.red[row][ch][0] = s; sh.red[row][ch][1] = s2;
  __syncthreads();
  if (tid < 16) {
    float S = 0.f, S2 = 0.f;
    for (int c2 = 0; c2 < 8; ++c2) { S += sh.red[tid][c2][0]; S2 += sh.red[tid][c2][1]; }
    float mean = S / dim;
    float var  = S2 / dim - mean * mean;
    sh.mstats[tid][0] = mean;
    sh.mstats[tid][1] = rsqrtf(var + 1e-5f);
  }
  __syncthreads();
  float mean = sh.mstats[row][0], inv = sh.mstats[row][1];
  _Float16* op = out + row * outPitch;
  for (int j = ch * per; j < (ch + 1) * per; ++j) {
    float x = ((float)ip[j] - mean) * inv * g[j] + bta[j];
    if (relu) x = fmaxf(x, 0.f);
    op[j] = (_Float16)x;
  }
  __syncthreads();
}

__global__ __launch_bounds__(128) void rqvae_fused(
    const float* __restrict__ semIn, const float* __restrict__ colIn,
    const float* __restrict__ gate_b1, const float* __restrict__ gate_ln_g,
    const float* __restrict__ gate_ln_b, const float* __restrict__ gate_b2,
    const float* __restrict__ fus_g, const float* __restrict__ fus_b,
    const float* __restrict__ sem_b1, const float* __restrict__ sem_ln_g,
    const float* __restrict__ sem_ln_b, const float* __restrict__ sem_b2,
    const float* __restrict__ col_b1, const float* __restrict__ col_ln_g,
    const float* __restrict__ col_ln_b, const float* __restrict__ col_b2,
    const float* __restrict__ cbf, const _Float16* __restrict__ wsH,
    const float* __restrict__ eh, float* __restrict__ dout) {
  __shared__ SMem sh;
  int tid = threadIdx.x, lane = tid & 31, waveId = tid >> 5;
  long long rowBase = (long long)blockIdx.x * 16;

  // ---- stage inputs (vectorized): semantic f16 -> bufA[:,0:768]
  for (int i = tid; i < 16 * 192; i += 128) {            // 192 float4 per row
    int m = i / 192, c = (i - m * 192) * 4;
    f32x4 v = *reinterpret_cast<const f32x4*>(semIn + (rowBase + m) * 768 + c);
    union { _Float16 h[4]; double d; } u;
#pragma unroll
    for (int t = 0; t < 4; ++t) u.h[t] = (_Float16)v[t];
    *reinterpret_cast<double*>(&sh.bufA[m][c]) = u.d;
  }
  // collab f32 -> small, f16 -> bufB[:,0:64]
  for (int i = tid; i < 16 * 16; i += 128) {             // 16 float4 per row
    int m = i >> 4, c = (i & 15) * 4;
    f32x4 v = *reinterpret_cast<const f32x4*>(colIn + (rowBase + m) * 64 + c);
    *reinterpret_cast<f32x4*>(&sh.small[m][c]) = v;
    union { _Float16 h[4]; double d; } u;
#pragma unroll
    for (int t = 0; t < 4; ++t) u.h[t] = (_Float16)v[t];
    *reinterpret_cast<double*>(&sh.bufB[m][c]) = u.d;
  }
  if (tid == 0) sh.loss = 0.f;
  for (int i = tid; i < 16 * 64; i += 128) sh.zq[i >> 6][i & 63] = 0.f;
  __syncthreads();

  // ---- gate MLP: [16,64] -> 128 (raw->LN->ReLU) -> 64 (sigmoid)
  gemm_stage(&sh.bufB[0][0], PITCH_B, 64, wsH + W_GATE1, 128, gate_b1, 0,
             &sh.bufB[0][128], PITCH_B, nullptr, 0, nullptr, 0, 0, waveId, lane);
  __syncthreads();
  ln_block(sh, &sh.bufB[0][128], PITCH_B, 128, gate_ln_g, gate_ln_b,
           &sh.bufB[0][256], PITCH_B, true, tid);
  { // gate GEMM2 + sigmoid -> gate_values (global) and denoised -> bufA[:,768:832]
    const _Float16* Wg2 = wsH + W_GATE2;
    for (int nt = waveId; nt < 4; nt += NWAVES) {
      int nB = nt << 4;
      v8f c = {};
      for (int kb = 0; kb < 128; kb += 32) {
        v16h a = ld_a(&sh.bufB[0][256], PITCH_B, kb, lane);
        v16h b = ld_b(Wg2, 64, kb, nB, lane);
        c = __builtin_amdgcn_wmma_f32_16x16x32_f16(false, a, false, b, (short)0, c, false, false);
      }
      int n = lane & 15, hf = lane >> 4;
      float bv = gate_b2[nB + n];
#pragma unroll
      for (int r = 0; r < 8; ++r) {
        int m = r + (hf << 3);
        float gv = 1.0f / (1.0f + __expf(-(c[r] + bv)));
        dout[O_GATE + (rowBase + m) * 64 + nB + n] = gv;
        sh.bufA[m][768 + nB + n] = (_Float16)(gv * sh.small[m][nB + n]);
      }
    }
  }
  __syncthreads();

  // ---- fusion LN over 832: bufA -> bufB
  ln_block(sh, &sh.bufA[0][0], PITCH_A, 832, fus_g, fus_b, &sh.bufB[0][0], PITCH_B, false, tid);

  // ---- encoder 832->512->256->128->64
  gemm_stage(&sh.bufB[0][0], PITCH_B, 832, wsH + W_ENC1, 512, nullptr, 1,
             &sh.bufA[0][0], PITCH_A, nullptr, 0, nullptr, 0, 0, waveId, lane);
  __syncthreads();
  gemm_stage(&sh.bufA[0][0], PITCH_A, 512, wsH + W_ENC2, 256, nullptr, 1,
             &sh.bufB[0][0], PITCH_B, nullptr, 0, nullptr, 0, 0, waveId, lane);
  __syncthreads();
  gemm_stage(&sh.bufB[0][0], PITCH_B, 256, wsH + W_ENC3, 128, nullptr, 1,
             &sh.bufA[0][0], PITCH_A, nullptr, 0, nullptr, 0, 0, waveId, lane);
  __syncthreads();
  gemm_stage(&sh.bufA[0][0], PITCH_A, 128, wsH + W_ENC4, 64, nullptr, 4,
             nullptr, 0, &sh.small[0][0], 68, nullptr, 0, 0, waveId, lane);
  __syncthreads();

  // ---- residual quantization, 3 layers
  for (int l = 0; l < 3; ++l) {
    // l2norm rows of small -> resn (f32) and bufB[:,0:64] (f16)
    {
      int row = tid & 15, ch = tid >> 4;
      float s = 0.f;
      for (int j = ch * 8; j < ch * 8 + 8; ++j) { float x = sh.small[row][j]; s += x * x; }
      sh.red[row][ch][0] = s;
      __syncthreads();
      if (tid < 16) {
        float S = 0.f;
        for (int c2 = 0; c2 < 8; ++c2) S += sh.red[tid][c2][0];
        sh.mstats[tid][0] = 1.0f / fmaxf(sqrtf(S), 1e-12f);
      }
      __syncthreads();
      float sc = sh.mstats[row][0];
      for (int j = ch * 8; j < ch * 8 + 8; ++j) {
        float v = sh.small[row][j] * sc;
        sh.resn[row][j] = v;
        sh.bufB[row][j] = (_Float16)v;
      }
      __syncthreads();
    }
    // scores: resn[16,64] x cbT[64,256], dual-tile WMMA + fused argmax (low-index tie-break)
    {
      const _Float16* cbT = wsH + W_CBT + (size_t)l * 64 * 256;
      float bestV[8]; int bestI[8];
#pragma unroll
      for (int r = 0; r < 8; ++r) { bestV[r] = -3.4e38f; bestI[r] = 0; }
      for (int j = 0; j < 2; ++j) {                  // 8 tile-pairs over 4 waves
        int nB0 = (waveId * 2 + j) << 5, nB1 = nB0 + 16;
        v8f c0 = {}, c1 = {};
        for (int kb = 0; kb < 64; kb += 32) {
          v16h a  = ld_a(&sh.bufB[0][0], PITCH_B, kb, lane);
          v16h b0 = ld_b(cbT, 256, kb, nB0, lane);
          v16h b1 = ld_b(cbT, 256, kb, nB1, lane);
          c0 = __builtin_amdgcn_wmma_f32_16x16x32_f16(false, a, false, b0, (short)0, c0, false, false);
          c1 = __builtin_amdgcn_wmma_f32_16x16x32_f16(false, a, false, b1, (short)0, c1, false, false);
        }
        int n = lane & 15;
        float eh0 = eh[l * 256 + nB0 + n];
        float eh1 = eh[l * 256 + nB1 + n];
#pragma unroll
        for (int r = 0; r < 8; ++r) {
          float s0 = c0[r] - eh0;
          int   i0 = nB0 + n;
          if (s0 > bestV[r] || (s0 == bestV[r] && i0 < bestI[r])) { bestV[r] = s0; bestI[r] = i0; }
          float s1 = c1[r] - eh1;
          int   i1 = nB1 + n;
          if (s1 > bestV[r] || (s1 == bestV[r] && i1 < bestI[r])) { bestV[r] = s1; bestI[r] = i1; }
        }
      }
      for (int off = 1; off < 16; off <<= 1) {
#pragma unroll
        for (int r = 0; r < 8; ++r) {
          float ov = __shfl_xor(bestV[r], off);
          int   oi = __shfl_xor(bestI[r], off);
          if (ov > bestV[r] || (ov == bestV[r] && oi < bestI[r])) { bestV[r] = ov; bestI[r] = oi; }
        }
      }
      int hf = lane >> 4;
      if ((lane & 15) == 0) {
#pragma unroll
        for (int r = 0; r < 8; ++r) {
          sh.candV[waveId][r + (hf << 3)] = bestV[r];
          sh.candI[waveId][r + (hf << 3)] = bestI[r];
        }
      }
      __syncthreads();
      if (tid < 16) {
        float bv = sh.candV[0][tid]; int bi = sh.candI[0][tid];
        for (int w = 1; w < NWAVES; ++w) {
          float v = sh.candV[w][tid]; int i2 = sh.candI[w][tid];
          if (v > bv || (v == bv && i2 < bi)) { bv = v; bi = i2; }
        }
        sh.codes[tid] = bi;
        reinterpret_cast<int*>(dout)[O_CODES + (rowBase + tid) * 3 + l] = bi;
      }
      __syncthreads();
    }
    // gather e, losses, zq accumulate, next residual
    {
      float lp = 0.f;
      for (int i = tid; i < 16 * 64; i += 128) {
        int m = i >> 6, k = i & 63;
        float e = cbf[((size_t)l * 256 + sh.codes[m]) * 64 + k];
        float r = sh.resn[m][k];
        float d = e - r;
        lp += d * d;
        sh.zq[m][k] += e;
        sh.small[m][k] = r - e;
      }
      atomicAdd(&sh.loss, lp);
      __syncthreads();
    }
  }

  // ---- z_q_total out + f16 for decoder
  for (int i = tid; i < 16 * 64; i += 128) {
    int m = i >> 6, k = i & 63;
    float z = sh.zq[m][k];
    dout[O_ZQ + (rowBase + m) * 64 + k] = z;
    sh.bufA[m][k] = (_Float16)z;
  }
  __syncthreads();

  // ---- shared decoder 64->128->256->512 (SiLU)
  gemm_stage(&sh.bufA[0][0], PITCH_A, 64, wsH + W_DEC1, 128, nullptr, 1,
             &sh.bufB[0][0], PITCH_B, nullptr, 0, nullptr, 0, 0, waveId, lane);
  __syncthreads();
  gemm_stage(&sh.bufB[0][0], PITCH_B, 128, wsH + W_DEC2, 256, nullptr, 1,
             &sh.bufA[0][0], PITCH_A, nullptr, 0, nullptr, 0, 0, waveId, lane);
  __syncthreads();
  gemm_stage(&sh.bufA[0][0], PITCH_A, 256, wsH + W_DEC3, 512, nullptr, 1,
             &sh.bufB[0][0], PITCH_B, nullptr, 0, nullptr, 0, 0, waveId, lane);
  __syncthreads();

  // ---- semantic head: 512->1536 (raw) -> LN+ReLU (in place) -> 768 (global)
  gemm_stage(&sh.bufB[0][0], PITCH_B, 512, wsH + W_SEM1, 1536, sem_b1, 0,
             &sh.bufA[0][0], PITCH_A, nullptr, 0, nullptr, 0, 0, waveId, lane);
  __syncthreads();
  ln_block(sh, &sh.bufA[0][0], PITCH_A, 1536, sem_ln_g, sem_ln_b,
           &sh.bufA[0][0], PITCH_A, true, tid);
  gemm_stage(&sh.bufA[0][0], PITCH_A, 1536, wsH + W_SEM2, 768, sem_b2, 3,
             nullptr, 0, nullptr, 0, dout + O_SEM, 768, rowBase, waveId, lane);
  __syncthreads();

  // ---- collab head: 512->256 (raw) -> LN+ReLU -> 64 (global)
  gemm_stage(&sh.bufB[0][0], PITCH_B, 512, wsH + W_COL1, 256, col_b1, 0,
             &sh.bufA[0][0], PITCH_A, nullptr, 0, nullptr, 0, 0, waveId, lane);
  __syncthreads();
  ln_block(sh, &sh.bufA[0][0], PITCH_A, 256, col_ln_g, col_ln_b,
           &sh.bufA[0][0], PITCH_A, true, tid);
  gemm_stage(&sh.bufA[0][0], PITCH_A, 256, wsH + W_COL2, 64, col_b2, 3,
             nullptr, 0, nullptr, 0, dout + O_COL, 64, rowBase, waveId, lane);

  if (tid == 0) {
    float v = sh.loss * (1.0f / ((float)B_ROWS * 64.0f));
    atomicAdd(dout + O_LOSS, v);       // cb_loss (forward value == cm_loss)
    atomicAdd(dout + O_LOSS + 1, v);   // cm_loss
  }
}

// ---------------- prepass kernels ----------------
__global__ void cvt_f32_f16(const float* __restrict__ src, _Float16* __restrict__ dst, int n) {
  int i = blockIdx.x * 256 + threadIdx.x;
  if (i < n) dst[i] = (_Float16)src[i];
}
__global__ void build_cbt(const float* __restrict__ cb, _Float16* __restrict__ cbT) {
  int i = blockIdx.x * 256 + threadIdx.x;           // over 3*256*64
  if (i < 3 * 256 * 64) {
    int l = i / (256 * 64), n = (i / 64) % 256, k = i % 64;
    cbT[((size_t)l * 64 + k) * 256 + n] = (_Float16)cb[i];
  }
}
__global__ void build_eh(const float* __restrict__ cb, float* __restrict__ eh) {
  int i = blockIdx.x * 256 + threadIdx.x;           // 768 codes
  if (i < 768) {
    float s = 0.f;
    for (int k = 0; k < 64; ++k) { float v = cb[(size_t)i * 64 + k]; s += v * v; }
    eh[i] = 0.5f * s;
  }
}
__global__ void zero_losses(float* p) { if (threadIdx.x < 2) p[threadIdx.x] = 0.f; }

extern "C" void kernel_launch(void* const* d_in, const int* in_sizes, int n_in,
                              void* d_out, int out_size, void* d_ws, size_t ws_size,
                              hipStream_t stream) {
  (void)in_sizes; (void)n_in; (void)out_size; (void)ws_size;
  _Float16* wsH = (_Float16*)d_ws;
  float* eh = (float*)((char*)d_ws + EH_BYTE);
  float* dout = (float*)d_out;

  auto cvt = [&](int idx, size_t off, int n) {
    cvt_f32_f16<<<(n + 255) / 256, 256, 0, stream>>>((const float*)d_in[idx], wsH + off, n);
  };
  cvt(2,  W_GATE1, 64 * 128);
  cvt(6,  W_GATE2, 128 * 64);
  cvt(10, W_ENC1,  832 * 512);
  cvt(11, W_ENC2,  512 * 256);
  cvt(12, W_ENC3,  256 * 128);
  cvt(13, W_ENC4,  128 * 64);
  cvt(14, W_DEC1,  64 * 128);
  cvt(15, W_DEC2,  128 * 256);
  cvt(16, W_DEC3,  256 * 512);
  cvt(17, W_SEM1,  512 * 1536);
  cvt(21, W_SEM2,  1536 * 768);
  cvt(23, W_COL1,  512 * 256);
  cvt(27, W_COL2,  256 * 64);
  build_cbt<<<(3 * 256 * 64 + 255) / 256, 256, 0, stream>>>((const float*)d_in[29], wsH + W_CBT);
  build_eh<<<3, 256, 0, stream>>>((const float*)d_in[29], eh);
  zero_losses<<<1, 32, 0, stream>>>(dout + O_LOSS);

  rqvae_fused<<<B_ROWS / 16, 128, 0, stream>>>(
      (const float*)d_in[0],  (const float*)d_in[1],
      (const float*)d_in[3],  (const float*)d_in[4],  (const float*)d_in[5],
      (const float*)d_in[7],
      (const float*)d_in[8],  (const float*)d_in[9],
      (const float*)d_in[18], (const float*)d_in[19], (const float*)d_in[20],
      (const float*)d_in[22],
      (const float*)d_in[24], (const float*)d_in[25], (const float*)d_in[26],
      (const float*)d_in[28],
      (const float*)d_in[29], wsH, eh, dout);
}
